// GCNBranch_33663953666142
// MI455X (gfx1250) — matrix operand order repaired and checked by reading, not compiled, not run on hardware
//
#include <hip/hip_runtime.h>
#include <math.h>

typedef _Float16 f16;
typedef __attribute__((ext_vector_type(16))) _Float16 v16h;
typedef __attribute__((ext_vector_type(8)))  float    v8f;

#define Bn    4
#define Cch   5
#define Hh    768
#define Ww    768
#define HW    589824          // 768*768
#define Nn    2048
#define HID   64
#define Kn    8
#define AB_W  64              // N/32 words per adjacency row

union FragU { v16h v; unsigned int u[8]; };

// ---- WMMA fragment loaders (layouts per CDNA5 ISA 7.12.2) -------------------

// A matrix 16x32 f16, row-major source with leading dim ld.
// lane: m = lane&15, g = lane>>4 ; element e -> k = 8g + 2*(e&3) + 16*(e>>2)
__device__ __forceinline__ v16h load_a_rm(const f16* tile, int ld, int lane) {
  FragU r; int m = lane & 15, g = lane >> 4;
  const f16* p = tile + m * ld + 8 * g;
#pragma unroll
  for (int e = 0; e < 8; ++e)
    r.u[e] = *reinterpret_cast<const unsigned int*>(p + 2 * (e & 3) + 16 * (e >> 2));
  return r.v;
}

// A matrix 16x32 f16 from LDS tile[16][32]
__device__ __forceinline__ v16h load_a_lds(const f16* tile, int lane) {
  FragU r; int m = lane & 15, g = lane >> 4;
  const f16* p = tile + m * 32 + 8 * g;
#pragma unroll
  for (int e = 0; e < 8; ++e)
    r.u[e] = *reinterpret_cast<const unsigned int*>(p + 2 * (e & 3) + 16 * (e >> 2));
  return r.v;
}

// B matrix 32x16 f16: B[kk][n] = colbase[n*stride + kk], kk = 8g + 2*(e&3) + 16*(e>>2) + h
__device__ __forceinline__ v16h load_b_strided(const f16* colbase, int stride, int lane) {
  FragU r; int n = lane & 15, g = lane >> 4;
  const f16* p = colbase + n * stride + 8 * g;
#pragma unroll
  for (int e = 0; e < 8; ++e)
    r.u[e] = *reinterpret_cast<const unsigned int*>(p + 2 * (e & 3) + 16 * (e >> 2));
  return r.v;
}

__device__ __forceinline__ v8f wmma_f16(v16h a, v16h b, v8f c) {
  return __builtin_amdgcn_wmma_f32_16x16x32_f16(false, a, false, b, (short)0, c, false, false);
}

// ---- kernels ----------------------------------------------------------------

__global__ __launch_bounds__(256) void k_zero_u32(unsigned int* p, int n) {
  int t = blockIdx.x * 256 + threadIdx.x;
  if (t < n) p[t] = 0u;
}

// LDS-staged segment accumulation: 8 slots per node {f0..f4, sum_y, sum_x, cnt}
__global__ __launch_bounds__(256) void k_seg(const float* __restrict__ x,
                                             const int* __restrict__ spx,
                                             float* __restrict__ acc) {
  __shared__ float s[Nn * 8];            // 64 KB of the 320 KB WGP LDS
  const int tid = threadIdx.x;
  const int b = blockIdx.x / 36, blk = blockIdx.x % 36;   // 36 blocks * 16384 px = HW
  for (int i = tid; i < Nn * 8; i += 256) s[i] = 0.f;
  __syncthreads();
  const int p0 = blk * 16384;
  const int* sp = spx + b * HW;
  const float* xb = x + (size_t)b * Cch * HW;
#pragma unroll 4
  for (int it = 0; it < 64; ++it) {
    int p = p0 + tid + it * 256;
    int id = sp[p];
    float* sa = s + id * 8;
    atomicAdd(sa + 0, xb[p]);
    atomicAdd(sa + 1, xb[HW + p]);
    atomicAdd(sa + 2, xb[2 * HW + p]);
    atomicAdd(sa + 3, xb[3 * HW + p]);
    atomicAdd(sa + 4, xb[4 * HW + p]);
    atomicAdd(sa + 5, (float)(p / Ww));
    atomicAdd(sa + 6, (float)(p % Ww));
    atomicAdd(sa + 7, 1.0f);
  }
  __syncthreads();
  float* ga = acc + (size_t)b * Nn * 8;
  for (int i = tid; i < Nn * 8; i += 256) {
    float v = s[i];
    if (v != 0.f) atomicAdd(ga + i, v);
  }
}

__global__ __launch_bounds__(256) void k_finalize(const float* __restrict__ acc,
                                                  float* __restrict__ node,
                                                  float* __restrict__ pos) {
  int t = blockIdx.x * 256 + threadIdx.x;          // B*N
  if (t >= Bn * Nn) return;
  const float* a = acc + (size_t)t * 8;
  float cnt = fmaxf(a[7], 1.0f);
  float inv = 1.0f / cnt;
#pragma unroll
  for (int c = 0; c < Cch; ++c) node[(size_t)t * Cch + c] = a[c] * inv;
  pos[(size_t)t * 2 + 0] = a[6] * inv * (1.0f / (float)(Ww - 1));  // cx
  pos[(size_t)t * 2 + 1] = a[5] * inv * (1.0f / (float)(Hh - 1));  // cy
}

__global__ __launch_bounds__(256) void k_knn(const float* __restrict__ pos,
                                             int* __restrict__ knn) {
  int t = blockIdx.x * 256 + threadIdx.x;          // B*N
  if (t >= Bn * Nn) return;
  int b = t >> 11, i = t & (Nn - 1);
  const float* pb = pos + (size_t)b * Nn * 2;
  float xi = pb[i * 2], yi = pb[i * 2 + 1];
  float bd[Kn]; int bj[Kn];
#pragma unroll
  for (int k = 0; k < Kn; ++k) { bd[k] = 3.4e38f; bj[k] = 0; }
  for (int j = 0; j < Nn; ++j) {
    if (j == i) continue;
    float dx = pb[j * 2] - xi, dy = pb[j * 2 + 1] - yi;
    float d = dx * dx + dy * dy;                 // monotone with sqrt distance
    if (d < bd[Kn - 1]) {
      int k = Kn - 1;
      while (k > 0 && bd[k - 1] > d) { bd[k] = bd[k - 1]; bj[k] = bj[k - 1]; --k; }
      bd[k] = d; bj[k] = j;                      // strict '>' keeps earlier index on ties
    }
  }
#pragma unroll
  for (int k = 0; k < Kn; ++k) knn[(size_t)t * Kn + k] = bj[k];
}

__global__ __launch_bounds__(256) void k_adj(const int* __restrict__ knn,
                                             unsigned int* __restrict__ ab) {
  int t = blockIdx.x * 256 + threadIdx.x;          // B*N
  if (t >= Bn * Nn) return;
#pragma unroll
  for (int k = 0; k < Kn; ++k) {
    int j = knn[(size_t)t * Kn + k];
    atomicOr(&ab[(size_t)t * AB_W + (j >> 5)], 1u << (j & 31));
  }
}

__global__ __launch_bounds__(256) void k_nedges(const int* __restrict__ knn,
                                                const unsigned int* __restrict__ ab,
                                                float* __restrict__ ne) {
  int t = blockIdx.x * 256 + threadIdx.x;          // B*N
  if (t >= Bn * Nn) return;
  int b = t >> 11, i = t & (Nn - 1);
  const unsigned int* A = ab + (size_t)b * Nn * AB_W;
  int cnt = 0;
#pragma unroll
  for (int k = 0; k < Kn; ++k) {
    int j = knn[(size_t)t * Kn + k];
    cnt += (int)((A[(size_t)j * AB_W + (i >> 5)] >> (i & 31)) & 1u);
  }
  atomicAdd(&ne[b], (float)cnt);
}

// Q/K/V (+optional res) projections; writes f16 Q,K row-major and f16 V transposed.
__global__ __launch_bounds__(256) void k_proj(const float* __restrict__ hin, int cin,
    const float* __restrict__ qw, const float* __restrict__ qb,
    const float* __restrict__ kw, const float* __restrict__ kb,
    const float* __restrict__ vw, const float* __restrict__ vb,
    const float* __restrict__ resw,
    f16* __restrict__ qf, f16* __restrict__ kf, f16* __restrict__ vt,
    float* __restrict__ resout) {
  int t = blockIdx.x * 256 + threadIdx.x;          // B*N*64
  int f = t & 63; int nb = t >> 6;                 // nb = b*N + n
  int b = nb >> 11; int nrow = nb & (Nn - 1);
  const float* hr = hin + (size_t)nb * cin;
  float q = qb[f], k = kb[f], v = vb[f], r = 0.f;
  for (int c = 0; c < cin; ++c) {
    float hv = hr[c];
    q += hv * qw[f * cin + c];
    k += hv * kw[f * cin + c];
    v += hv * vw[f * cin + c];
    if (resw) r += hv * resw[f * cin + c];
  }
  qf[t] = (f16)q; kf[t] = (f16)k;
  vt[((size_t)b * HID + f) * Nn + nrow] = (f16)v;
  if (resw) resout[t] = r;
}

// Fused masked attention: one wave per 16-row tile; WMMA QK^T + WMMA G*V,
// LDS-cached mutual-kNN bitset, post-loop row normalization, residual+LN+ReLU.
__global__ __launch_bounds__(32) void k_attn(const f16* __restrict__ qf,
                                             const f16* __restrict__ kf,
                                             const f16* __restrict__ vt,
                                             const float* __restrict__ res,
                                             const unsigned int* __restrict__ ab,
                                             const float* __restrict__ taup,
                                             const float* __restrict__ lng,
                                             const float* __restrict__ lnb,
                                             float* __restrict__ hout,
                                             float* __restrict__ gsum) {
  const int lane = threadIdx.x;
  const int b = blockIdx.y;
  const int row0 = blockIdx.x * 16;
  const f16* Q  = qf + (size_t)b * Nn * HID;
  const f16* Kf = kf + (size_t)b * Nn * HID;
  const f16* Vt = vt + (size_t)b * HID * Nn;
  const unsigned int* A = ab + (size_t)b * Nn * AB_W;
  const float tauv = taup[0];

  __shared__ unsigned int smask[16 * AB_W];  // bitset rows of this tile's 16 rows
  __shared__ unsigned int scol[Nn];          // transposed word A[j][row0>>5] for all j
  __shared__ f16  gt[16 * 32];
  __shared__ float sden[16];
  __shared__ float sg;
  __shared__ float htmp[16 * HID];

  // preload mask words into LDS (removes all global mask traffic from hot loop)
  for (int idx = lane; idx < 16 * AB_W; idx += 32)
    smask[idx] = A[(size_t)(row0 + (idx >> 6)) * AB_W + (idx & 63)];
  const int cw = row0 >> 5;
  for (int idx = lane; idx < Nn; idx += 32)
    scol[idx] = A[(size_t)idx * AB_W + cw];
  if (lane < 16) sden[lane] = 0.f;
  if (lane == 0) sg = 0.f;
  __syncthreads();

  const int n = lane & 15, gg = lane >> 4;
  const int ib = row0 & 31;                  // base bit for transposed lookups
  // Q tile as two A fragments (K-dim 64 = 2 x 32), resident whole kernel
  v16h qa0 = load_a_rm(Q + (size_t)row0 * HID + 0,  HID, lane);
  v16h qa1 = load_a_rm(Q + (size_t)row0 * HID + 32, HID, lane);

  v8f accv[4]; accv[0] = {}; accv[1] = {}; accv[2] = {}; accv[3] = {};
  float denp[8] = {0, 0, 0, 0, 0, 0, 0, 0};
  float gpart = 0.f;

  for (int ct = 0; ct < Nn / 32; ++ct) {
    const int col0 = ct * 32;
    // ---- scores: two 16x16 tiles over 32 columns ----
    v8f sc[2];
#pragma unroll
    for (int nt = 0; nt < 2; ++nt) {
      const f16* cb = Kf + (size_t)(col0 + nt * 16) * HID;
      v16h b0 = load_b_strided(cb + 0,  HID, lane);
      v16h b1 = load_b_strided(cb + 32, HID, lane);
      v8f c = {};
      c = wmma_f16(qa0, b0, c);
      c = wmma_f16(qa1, b1, c);
      sc[nt] = c;
    }
    // ---- gate + mask + row-sum partials; stage g tile to LDS as f16 ----
    unsigned int mrow[8];
#pragma unroll
    for (int r = 0; r < 8; ++r) mrow[r] = smask[(r + 8 * gg) * AB_W + ct];
#pragma unroll
    for (int nt = 0; nt < 2; ++nt) {
      v8f sv = sc[nt];
      const int jb = nt * 16 + n;            // j&31 (col0 is a multiple of 32)
      const unsigned int cwrd = scol[col0 + jb];
#pragma unroll
      for (int r = 0; r < 8; ++r) {
        int m = r + 8 * gg;
        float e = sv[r] * 0.125f;                       // / sqrt(64)
        float gv = 1.0f / (1.0f + __expf(-(e - tauv) * (1.0f / 0.7f)));
        unsigned int ok = (mrow[r] >> jb) & (cwrd >> (ib + m)) & 1u;
        if (!ok) gv = 0.f;
        denp[r] += gv;
        gpart   += gv;
        gt[m * 32 + jb] = (f16)gv;
      }
    }
    __syncthreads();
    v16h ga = load_a_lds(gt, lane);
    __syncthreads();
    // ---- aggregation: AGG[16][64] += G(16x32) @ V(32x64) ----
#pragma unroll
    for (int nt = 0; nt < 4; ++nt) {
      v16h vb = load_b_strided(Vt + (size_t)(nt * 16) * Nn + col0, Nn, lane);
      accv[nt] = wmma_f16(ga, vb, accv[nt]);
    }
  }

  // reduce row sums / g-sum across the wave
#pragma unroll
  for (int r = 0; r < 8; ++r) atomicAdd(&sden[r + 8 * gg], denp[r]);
  atomicAdd(&sg, gpart);
  __syncthreads();
  if (lane == 0) atomicAdd(&gsum[b], sg);

  // spill AGG (C-layout) to LDS for the LayerNorm phase
#pragma unroll
  for (int nt = 0; nt < 4; ++nt)
#pragma unroll
    for (int r = 0; r < 8; ++r)
      htmp[(r + 8 * gg) * HID + nt * 16 + n] = accv[nt][r];
  __syncthreads();

  if (lane < 16) {
    const int row = lane, i = row0 + row;
    const float den = fmaxf(sden[row], 1e-6f);
    const float* rr = res + ((size_t)b * Nn + i) * HID;
    float* hrow = htmp + row * HID;
#pragma unroll
    for (int f = 0; f < HID; ++f) hrow[f] = rr[f] + hrow[f] / den;
    float mean = 0.f;
#pragma unroll
    for (int f = 0; f < HID; ++f) mean += hrow[f];
    mean *= (1.0f / HID);
    float var = 0.f;
#pragma unroll
    for (int f = 0; f < HID; ++f) { float d = hrow[f] - mean; var += d * d; }
    var *= (1.0f / HID);
    const float rstd = rsqrtf(var + 1e-5f);
    float* op = hout + ((size_t)b * Nn + i) * HID;
#pragma unroll
    for (int f = 0; f < HID; ++f)
      op[f] = fmaxf((hrow[f] - mean) * rstd * lng[f] + lnb[f], 0.f);
  }
}

__global__ void k_reg(const float* __restrict__ gsum, const float* __restrict__ ne,
                      float* __restrict__ outreg) {
  if (blockIdx.x == 0 && threadIdx.x == 0) {
    float r = 0.f;
    for (int b = 0; b < Bn; ++b) {
      float nv = fmaxf(ne[b], 1.0f);
      r += fabsf(gsum[b] / (2.0f * nv) - 0.3f);
    }
    outreg[0] = r * (1.0f / Bn);
  }
}

// Broadcast node features to pixels; b128 gathers of the L2-resident node
// table, fully coalesced b32 stores (the 604 MB HBM-write roofline term).
__global__ __launch_bounds__(256) void k_bcast(const float* __restrict__ h2,
                                               const int* __restrict__ spx,
                                               float* __restrict__ out) {
  const int blk = blockIdx.x;                 // B * (HW/256)
  const int b = blk / (HW / 256), pb = blk % (HW / 256);
  const int p = pb * 256 + threadIdx.x;
  const int id = spx[(size_t)b * HW + p];
  const float4* hr = (const float4*)(h2 + ((size_t)b * Nn + id) * HID);  // 256B aligned
  float* ob = out + (size_t)b * HID * HW + p;
#pragma unroll
  for (int f4 = 0; f4 < HID / 4; ++f4) {
    float4 v = hr[f4];
    ob[(size_t)(4 * f4 + 0) * HW] = v.x;
    ob[(size_t)(4 * f4 + 1) * HW] = v.y;
    ob[(size_t)(4 * f4 + 2) * HW] = v.z;
    ob[(size_t)(4 * f4 + 3) * HW] = v.w;
  }
}

// ---- driver -----------------------------------------------------------------

extern "C" void kernel_launch(void* const* d_in, const int* in_sizes, int n_in,
                              void* d_out, int out_size, void* d_ws, size_t ws_size,
                              hipStream_t stream) {
  const float* x    = (const float*)d_in[0];
  const int*   spx  = (const int*)  d_in[1];
  const float* q1w  = (const float*)d_in[2];  const float* q1b = (const float*)d_in[3];
  const float* k1w  = (const float*)d_in[4];  const float* k1b = (const float*)d_in[5];
  const float* v1w  = (const float*)d_in[6];  const float* v1b = (const float*)d_in[7];
  const float* r1w  = (const float*)d_in[8];
  const float* ln1g = (const float*)d_in[9];  const float* ln1b = (const float*)d_in[10];
  const float* q2w  = (const float*)d_in[11]; const float* q2b = (const float*)d_in[12];
  const float* k2w  = (const float*)d_in[13]; const float* k2b = (const float*)d_in[14];
  const float* v2w  = (const float*)d_in[15]; const float* v2b = (const float*)d_in[16];
  const float* ln2g = (const float*)d_in[17]; const float* ln2b = (const float*)d_in[18];
  const float* tau  = (const float*)d_in[19];
  float* out = (float*)d_out;

  // workspace layout (zero-init regions first, contiguous)
  char* w = (char*)d_ws;
  float*        acc  = (float*)w;                                  // B*N*8 f32
  unsigned int* ab   = (unsigned int*)(w + (size_t)Bn * Nn * 8 * 4);  // B*N*64 u32
  float*        gsum = (float*)((char*)ab + (size_t)Bn * Nn * AB_W * 4); // 4
  float*        nedg = gsum + 4;                                   // 4
  float*        node = nedg + 4;                                   // B*N*5
  float*        pos  = node + (size_t)Bn * Nn * Cch;               // B*N*2
  int*          knn  = (int*)(pos + (size_t)Bn * Nn * 2);          // B*N*8
  f16*          qf   = (f16*)(knn + (size_t)Bn * Nn * Kn);         // B*N*64 f16
  f16*          kf   = qf + (size_t)Bn * Nn * HID;
  f16*          vt   = kf + (size_t)Bn * Nn * HID;
  float*        res  = (float*)(vt + (size_t)Bn * Nn * HID);       // B*N*64 f32
  float*        h1   = res + (size_t)Bn * Nn * HID;
  float*        h2   = h1  + (size_t)Bn * Nn * HID;

  const int zeroN = (Bn * Nn * 8 * 4 + Bn * Nn * AB_W * 4 + 32) / 4;
  k_zero_u32<<<(zeroN + 255) / 256, 256, 0, stream>>>((unsigned int*)w, zeroN);

  k_seg<<<Bn * 36, 256, 0, stream>>>(x, spx, acc);
  k_finalize<<<(Bn * Nn + 255) / 256, 256, 0, stream>>>(acc, node, pos);
  k_knn<<<(Bn * Nn + 255) / 256, 256, 0, stream>>>(pos, knn);
  k_adj<<<(Bn * Nn + 255) / 256, 256, 0, stream>>>(knn, ab);
  k_nedges<<<(Bn * Nn + 255) / 256, 256, 0, stream>>>(knn, ab, nedg);

  dim3 agrid(Nn / 16, Bn);
  // layer 1
  k_proj<<<(Bn * Nn * HID) / 256, 256, 0, stream>>>(node, Cch, q1w, q1b, k1w, k1b,
                                                    v1w, v1b, r1w, qf, kf, vt, res);
  k_attn<<<agrid, 32, 0, stream>>>(qf, kf, vt, res, ab, tau, ln1g, ln1b, h1, gsum);
  // layer 2 (residual = h1)
  k_proj<<<(Bn * Nn * HID) / 256, 256, 0, stream>>>(h1, HID, q2w, q2b, k2w, k2b,
                                                    v2w, v2b, nullptr, qf, kf, vt, res);
  k_attn<<<agrid, 32, 0, stream>>>(qf, kf, vt, h1, ab, tau, ln2g, ln2b, h2, gsum);

  k_reg<<<1, 32, 0, stream>>>(gsum, nedg, out + (size_t)Bn * HID * HW);
  k_bcast<<<Bn * (HW / 256), 256, 0, stream>>>(h2, spx, out);
}
